// AttnRNNDecModel_29635274342883
// MI455X (gfx1250) — matrix operand order repaired and checked by reading, not compiled, not run on hardware
//
#include <hip/hip_runtime.h>

typedef unsigned short u16;
typedef __attribute__((ext_vector_type(16))) __bf16 v16bf;
typedef __attribute__((ext_vector_type(8)))  float  v8f;
typedef unsigned int u32x4 __attribute__((ext_vector_type(4)));
typedef int          i32x8 __attribute__((ext_vector_type(8)));
typedef int          i32x4 __attribute__((ext_vector_type(4)));

#if defined(__has_builtin)
#  if __has_builtin(__builtin_amdgcn_tensor_load_to_lds) && __has_builtin(__builtin_amdgcn_s_wait_tensorcnt)
#    define USE_TDM 1
#  endif
#endif
#ifndef USE_TDM
#  define USE_TDM 0
#endif

union ABfrag  { u16 s[16]; v16bf v; };
union AccFrag { float f[8]; v8f  v; };

__device__ inline u16 f2b(float f) {
  unsigned u = __builtin_bit_cast(unsigned, f);
  unsigned r = u + 0x7FFFu + ((u >> 16) & 1u);   // round-to-nearest-even
  return (u16)(r >> 16);
}
__device__ inline float b2f(u16 h) {
  unsigned u = ((unsigned)h) << 16;
  return __builtin_bit_cast(float, u);
}

// ---------------- fp32 -> bf16 conversion ----------------
__global__ void f32_to_bf16_kernel(const float* __restrict__ in, u16* __restrict__ out, long n) {
  long i = (long)blockIdx.x * blockDim.x + threadIdx.x;
  long stride = (long)gridDim.x * blockDim.x;
  for (; i < n; i += stride) out[i] = f2b(in[i]);
}

// ---------------- generic bf16 WMMA GEMM:  C = act(A @ W^T + bias) ----------------
// A: (M,K) bf16 row-major (optionally row-gathered via gidx), W: (N,K) bf16 row-major.
// BM=64, BN=64, BK=32; 256 threads = 8 waves; each wave owns a 16x32 output strip
// (two 16x16 WMMA accumulators sharing one A fragment).
// W tile is staged into LDS by the Tensor Data Mover (wave 0 issues a 2-D D#
// with LDS padding so the DMA writes the bank-padded layout directly).
#define GBM 64
#define GBN 64
#define GBK 32
#define GLDA 40
#define GLDW 40

__global__ void __launch_bounds__(256)
gemm_bf16_wmma(const u16* __restrict__ A, const int* __restrict__ gidx,
               const u16* __restrict__ W, const float* __restrict__ bias,
               float* __restrict__ outF, u16* __restrict__ outB,
               int M, int N, int K, int relu)
{
  __shared__ u16 As[GBM * GLDA];
  __shared__ u16 Ws[GBN * GLDW];

  const int tid = threadIdx.x;
  const int lid = tid & 31;
  const int w   = tid >> 5;        // 0..7
  const int mt  = w & 3;           // 4 M-subtiles
  const int ntp = (w >> 2) * 2;    // first of two N-subtiles per wave
  const int hl  = lid >> 4;        // half-wave
  const int l16 = lid & 15;
  const int m0  = blockIdx.y * GBM;
  const int n0  = blockIdx.x * GBN;

  AccFrag acc0, acc1;
#pragma unroll
  for (int i = 0; i < 8; ++i) { acc0.f[i] = 0.f; acc1.f[i] = 0.f; }

  for (int k0 = 0; k0 < K; k0 += GBK) {
    // ---- stage A tile (64x32 bf16): 8 elems (16B) per thread
    {
      int row = tid >> 2;
      int cg  = (tid & 3) * 8;
      int gm  = m0 + row;
      uint4 val = {0u, 0u, 0u, 0u};
      if (gm < M) {
        long arow = gidx ? (long)gidx[gm] : (long)gm;
        const uint4* p = reinterpret_cast<const uint4*>(A + arow * (long)K + k0 + cg);
        val = *p;
        if (k0 + GBK < K) __builtin_prefetch(A + arow * (long)K + k0 + GBK + cg, 0, 1);
      }
      *reinterpret_cast<uint4*>(&As[row * GLDA + cg]) = val;
    }
    // ---- stage W tile (64 rows x 32 K) via Tensor Data Mover
#if USE_TDM
    if (w == 0) {
      unsigned lds_addr = (unsigned)(uintptr_t)(&Ws[0]);
      unsigned long long ga = (unsigned long long)(uintptr_t)(W + (long)n0 * K + k0);
      // D# group 0: count=1 | lds_addr | global_addr[56:0] | type=2
      u32x4 g0 = { 1u,
                   lds_addr,
                   (unsigned)(ga & 0xffffffffu),
                   (unsigned)((ga >> 32) & 0x01ffffffu) | 0x80000000u };
      // D# group 1: data_size=2B, pad every 64B by 16B (-> 40-elem LDS stride),
      // tensor (dim0=K, dim1=N, stride0=K), tile 32(K) x 64(N)
      i32x8 g1;
      g1[0] = (1 << 16) | (1 << 20) | (3 << 22) | (3 << 25);
      g1[1] = (int)(((unsigned)K & 0xffffu) << 16);
      g1[2] = (int)((((unsigned)K >> 16) & 0xffffu) | (((unsigned)N & 0xffffu) << 16));
      g1[3] = (int)((((unsigned)N >> 16) & 0xffffu) | (32u << 16));
      g1[4] = 64;              // tile_dim1=64, tile_dim2=0
      g1[5] = K;               // tensor_dim0_stride[31:0]
      g1[6] = 0;
      g1[7] = 0;
      i32x4 gz  = { 0, 0, 0, 0 };
      i32x8 gz8 = { 0, 0, 0, 0, 0, 0, 0, 0 };
      __builtin_amdgcn_tensor_load_to_lds(g0, g1, gz, gz, gz8, 0);
      __builtin_amdgcn_s_wait_tensorcnt(0);
    }
#else
    {
      int row = tid >> 2;
      int cg  = (tid & 3) * 8;
      const uint4* p = reinterpret_cast<const uint4*>(W + (long)(n0 + row) * K + k0 + cg);
      *reinterpret_cast<uint4*>(&Ws[row * GLDW + cg]) = *p;
    }
#endif
    __syncthreads();

    // ---- fragments + 2 WMMAs per wave
    ABfrag a, b0, b1;
    {
      const u16* ap = &As[(mt * 16 + l16) * GLDA + hl * 8];
#pragma unroll
      for (int j = 0; j < 8; ++j) a.s[j] = ap[j];          // K = hl*8 + j
#pragma unroll
      for (int j = 0; j < 8; ++j) a.s[8 + j] = ap[16 + j]; // K = 16 + hl*8 + j
      const u16* bp0 = &Ws[(ntp * 16 + l16) * GLDW + hl * 8];
      const u16* bp1 = &Ws[((ntp + 1) * 16 + l16) * GLDW + hl * 8];
#pragma unroll
      for (int e = 0; e < 16; ++e) { int o = e < 8 ? e : e + 8; b0.s[e] = bp0[o]; b1.s[e] = bp1[o]; }
    }
    acc0.v = __builtin_amdgcn_wmma_f32_16x16x32_bf16(false, a.v, false, b0.v,
                                                     (short)0, acc0.v, false, false);
    acc1.v = __builtin_amdgcn_wmma_f32_16x16x32_bf16(false, a.v, false, b1.v,
                                                     (short)0, acc1.v, false, false);
    __syncthreads();
  }

  // ---- epilogue
#pragma unroll
  for (int ti = 0; ti < 2; ++ti) {
    AccFrag& ac = ti ? acc1 : acc0;
    const int ng = n0 + (ntp + ti) * 16 + l16;
    const float bv = bias ? bias[ng] : 0.f;
#pragma unroll
    for (int v = 0; v < 8; ++v) {
      int mg = m0 + mt * 16 + v + hl * 8;
      if (mg < M) {
        float val = ac.f[v] + bv;
        if (relu) val = fmaxf(val, 0.f);
        long idx = (long)mg * N + ng;
        if (outF) outF[idx] = val;
        if (outB) outB[idx] = f2b(val);
      }
    }
  }
}

// ---------------- persistent attention-RNN layer ----------------
// One workgroup, 1024 threads = 32 waves. h kept in LDS as bf16.
// Per step: attention (wave per batch row), ctx+u, fused [u;h]@[Wih;Whh]^T via
// 32 wave-level 16x16 WMMA tiles (2 M-tiles x 16 N-tiles), tanh epilogue.
#define RB 32
#define RH 256
#define RS 128
#define RLD 264

__global__ void __launch_bounds__(1024)
rnn_attn_layer(const float* __restrict__ x_in,      // (B,S,H) fp32
               const float* __restrict__ h0,        // (B,H) fp32 (raw-reshape slice)
               const float* __restrict__ enc,       // (B,S,H) fp32
               const u16* __restrict__ Wih, const u16* __restrict__ Whh, // (H,H) bf16
               const float* __restrict__ bih, const float* __restrict__ bhh,
               float* __restrict__ xoF, u16* __restrict__ xoB)
{
  __shared__ float score[RB][RS];   // 16 KB
  __shared__ u16   u_bf[RB][RLD];   // ~16.5 KB
  __shared__ u16   h_bf[RB][RLD];   // ~16.5 KB

  const int tid = threadIdx.x;
  const int lid = tid & 31;
  const int w   = tid >> 5;         // 0..31
  const int hl  = lid >> 4;
  const int l16 = lid & 15;

  for (int i = tid; i < RB * RH; i += 1024) {
    int bb = i >> 8, hh = i & 255;
    h_bf[bb][hh] = f2b(h0[i]);
  }
  __syncthreads();

  for (int t = 0; t < RS; ++t) {
    // ---- scores: wave w <-> batch row b, 4 sequence positions per lane
    {
      const int b = w;
      const float* encb = enc + (long)b * RS * RH;
      float d[4];
#pragma unroll
      for (int q = 0; q < 4; ++q) {
        int s = q * 32 + lid;
        const float4* er = reinterpret_cast<const float4*>(encb + s * RH);
        float accs = 0.f;
        for (int hh = 0; hh < RH / 4; ++hh) {
          float4 e = er[hh];
          accs += b2f(h_bf[b][hh * 4 + 0]) * e.x + b2f(h_bf[b][hh * 4 + 1]) * e.y
                + b2f(h_bf[b][hh * 4 + 2]) * e.z + b2f(h_bf[b][hh * 4 + 3]) * e.w;
        }
        d[q] = accs;
      }
      float mx = fmaxf(fmaxf(d[0], d[1]), fmaxf(d[2], d[3]));
      for (int off = 16; off; off >>= 1) mx = fmaxf(mx, __shfl_xor(mx, off, 32));
      float sum = 0.f;
#pragma unroll
      for (int q = 0; q < 4; ++q) { d[q] = __expf(d[q] - mx); sum += d[q]; }
      for (int off = 16; off; off >>= 1) sum += __shfl_xor(sum, off, 32);
      float inv = 1.f / sum;
#pragma unroll
      for (int q = 0; q < 4; ++q) score[b][q * 32 + lid] = d[q] * inv;
    }
    __syncthreads();

    // ---- ctx + u = x_t + ctx, stored bf16 for WMMA A-operand
    {
      const int bb = tid >> 5;
      const int hbase = tid & 31;
      const float* encb = enc + (long)bb * RS * RH;
#pragma unroll
      for (int r = 0; r < 8; ++r) {
        int hh = hbase + r * 32;
        float c = 0.f;
        for (int s = 0; s < RS; ++s) c += score[bb][s] * encb[s * RH + hh];
        float uu = x_in[((long)bb * RS + t) * RH + hh] + c;
        u_bf[bb][hh] = f2b(uu);
      }
    }
    __syncthreads();

    // ---- fused GEMM + tanh: wave -> (mt, nt) 16x16 tile
    const int nt = w & 15, mt = w >> 4;
    const int ng = nt * 16 + l16;
    AccFrag acc;
#pragma unroll
    for (int i = 0; i < 8; ++i) acc.f[i] = 0.f;
    for (int pass = 0; pass < 2; ++pass) {
      const u16 (*Ax)[RLD] = pass ? (const u16 (*)[RLD])h_bf : (const u16 (*)[RLD])u_bf;
      const u16* Wx = pass ? Whh : Wih;
      for (int k0 = 0; k0 < RH; k0 += 32) {
        ABfrag a, b;
        const u16* ap = &Ax[mt * 16 + l16][k0 + hl * 8];
#pragma unroll
        for (int j = 0; j < 8; ++j) a.s[j] = ap[j];
#pragma unroll
        for (int j = 0; j < 8; ++j) a.s[8 + j] = ap[16 + j];
        const u16* bp = Wx + (long)ng * RH + k0 + hl * 8;
#pragma unroll
        for (int e = 0; e < 16; ++e) b.s[e] = bp[e < 8 ? e : e + 8];
        acc.v = __builtin_amdgcn_wmma_f32_16x16x32_bf16(false, a.v, false, b.v,
                                                        (short)0, acc.v, false, false);
      }
    }
    const float bsum = bih[ng] + bhh[ng];
    float hn[8];
#pragma unroll
    for (int v = 0; v < 8; ++v) hn[v] = tanhf(acc.f[v] + bsum);
    __syncthreads();   // all h_bf reads done before overwrite
#pragma unroll
    for (int v = 0; v < 8; ++v) {
      int mg = mt * 16 + v + hl * 8;
      u16 hb = f2b(hn[v]);
      h_bf[mg][ng] = hb;
      long idx = ((long)mg * RS + t) * RH + ng;
      xoF[idx] = hn[v];
      xoB[idx] = hb;
    }
    __syncthreads();
  }
}

// ---------------- host orchestration ----------------
extern "C" void kernel_launch(void* const* d_in, const int* in_sizes, int n_in,
                              void* d_out, int out_size, void* d_ws, size_t ws_size,
                              hipStream_t stream) {
  (void)in_sizes; (void)n_in; (void)out_size; (void)ws_size;
  const float* enc_hid  = (const float*)d_in[0];
  const float* enc_out  = (const float*)d_in[1];
  const int*   tgt      = (const int*)d_in[2];
  const float* emb_W    = (const float*)d_in[3];
  const float* e2h_b1   = (const float*)d_in[5];
  const float* e2h_b2   = (const float*)d_in[7];
  const float* enc2h_b1 = (const float*)d_in[9];
  const float* enc2h_b2 = (const float*)d_in[11];
  const float* rnn_bih  = (const float*)d_in[14];
  const float* rnn_bhh  = (const float*)d_in[15];
  const float* h2e_b1   = (const float*)d_in[17];
  const float* h2e_b2   = (const float*)d_in[19];

  const int V = 32000, E = 256, H = 256, L = 2, B = 32, S = 128, BS = B * S;

  char* ws = (char*)d_ws;
  size_t off = 0;
  auto alloc_b = [&](size_t bytes) -> void* {
    off = (off + 255) & ~(size_t)255;
    void* p = ws + off; off += bytes; return p;
  };
  u16*   embW_b = (u16*)alloc_b((size_t)V * E * 2);
  u16*   w_e2h1 = (u16*)alloc_b((size_t)H * E * 2);
  u16*   w_e2h2 = (u16*)alloc_b((size_t)H * H * 2);
  u16*   ench_b = (u16*)alloc_b((size_t)B * H * 2);
  u16*   w_enc1 = (u16*)alloc_b((size_t)L * H * H * 2);
  u16*   w_enc2 = (u16*)alloc_b((size_t)L * H * L * H * 2);
  u16*   w_ih   = (u16*)alloc_b((size_t)L * H * H * 2);
  u16*   w_hh   = (u16*)alloc_b((size_t)L * H * H * 2);
  u16*   w_h2e1 = (u16*)alloc_b((size_t)H * H * 2);
  u16*   w_h2e2 = (u16*)alloc_b((size_t)E * H * 2);
  u16*   h1_b   = (u16*)alloc_b((size_t)BS * H * 2);
  float* x0_f   = (float*)alloc_b((size_t)BS * H * 4);
  u16*   tenc_b = (u16*)alloc_b((size_t)B * L * H * 2);
  float* h0_raw = (float*)alloc_b((size_t)B * L * H * 4);
  float* x1_f   = (float*)alloc_b((size_t)BS * H * 4);
  u16*   x1_b   = (u16*)alloc_b((size_t)BS * H * 2);
  float* x2_f   = (float*)alloc_b((size_t)BS * H * 4);
  u16*   x2_b   = (u16*)alloc_b((size_t)BS * H * 2);
  u16*   t1_b   = (u16*)alloc_b((size_t)BS * H * 2);
  u16*   y_b    = (u16*)alloc_b((size_t)BS * E * 2);

  auto cvt = [&](const float* src, u16* dst, long n) {
    int blocks = (int)((n + 255) / 256);
    if (blocks > 2048) blocks = 2048;
    f32_to_bf16_kernel<<<blocks, 256, 0, stream>>>(src, dst, n);
  };
  cvt(emb_W,                  embW_b, (long)V * E);
  cvt((const float*)d_in[4],  w_e2h1, (long)H * E);
  cvt((const float*)d_in[6],  w_e2h2, (long)H * H);
  cvt(enc_hid,                ench_b, (long)B * H);
  cvt((const float*)d_in[8],  w_enc1, (long)L * H * H);
  cvt((const float*)d_in[10], w_enc2, (long)L * H * L * H);
  cvt((const float*)d_in[12], w_ih,   (long)L * H * H);
  cvt((const float*)d_in[13], w_hh,   (long)L * H * H);
  cvt((const float*)d_in[16], w_h2e1, (long)H * H);
  cvt((const float*)d_in[18], w_h2e2, (long)E * H);

  auto gemm = [&](const u16* A, const int* gidx, const u16* W, const float* bias,
                  float* oF, u16* oB, int M, int N, int K, int relu) {
    gemm_bf16_wmma<<<dim3(N / GBN, (M + GBM - 1) / GBM), 256, 0, stream>>>(
        A, gidx, W, bias, oF, oB, M, N, K, relu);
  };

  // emb + emb_to_hid (gathered A) -> x0
  gemm(embW_b, tgt,     w_e2h1, e2h_b1,   nullptr, h1_b, BS, H, E, 1);
  gemm(h1_b,   nullptr, w_e2h2, e2h_b2,   x0_f, nullptr, BS, H, H, 1);
  // enc_to_hid -> h0 (raw row-major reshape handled by flat slicing)
  gemm(ench_b, nullptr, w_enc1, enc2h_b1, nullptr, tenc_b, B, L * H, H, 1);
  gemm(tenc_b, nullptr, w_enc2, enc2h_b2, h0_raw, nullptr, B, L * H, L * H, 1);
  // two recurrent attention layers (persistent, one WGP each)
  rnn_attn_layer<<<1, 1024, 0, stream>>>(x0_f, h0_raw,         enc_out,
                                         w_ih,         w_hh,
                                         rnn_bih,      rnn_bhh, x1_f, x1_b);
  rnn_attn_layer<<<1, 1024, 0, stream>>>(x1_f, h0_raw + B * H, enc_out,
                                         w_ih + H * H, w_hh + H * H,
                                         rnn_bih + H,  rnn_bhh + H, x2_f, x2_b);
  // hid_to_emb
  gemm(x2_b, nullptr, w_h2e1, h2e_b1, nullptr, t1_b, BS, H, H, 1);
  gemm(t1_b, nullptr, w_h2e2, h2e_b2, nullptr, y_b,  BS, E, H, 0);
  // tied-embedding logits -> d_out (fp32)
  gemm(y_b,  nullptr, embW_b, nullptr, (float*)d_out, nullptr, BS, V, E, 0);
}
